// MHAttention_12206297055586
// MI455X (gfx1250) — matrix operand order repaired and checked by reading, not compiled, not run on hardware
//
#include <hip/hip_runtime.h>

// ---------------------------------------------------------------------------
// MI455X (gfx1250) linear-attention transformer block.
// All dense contractions use v_wmma_f32_16x16x32_f16 (wave32 WMMA, fp32 acc).
// A-tiles stream global->LDS via GLOBAL_LOAD_ASYNC_TO_LDS_B128 (ASYNCcnt),
// double-buffered so the async copy overlaps the WMMA stream.
// ---------------------------------------------------------------------------

typedef __attribute__((ext_vector_type(16))) _Float16 v16h;
typedef __attribute__((ext_vector_type(8)))  float    v8f;

#define D_MODEL 512
#define NHEAD   8
#define HDIM    64
#define SEQ     4096
#define BATCH   4
#define MROWS   (BATCH * SEQ)   // 16384

// ---- CDNA5 async global->LDS copy (guarded; falls back to sync staging) ----
#if __has_builtin(__builtin_amdgcn_global_load_async_to_lds_b128)
#define HAVE_ASYNC_COPY 1
// Builtin signature (from clang diagnostic): param 1 is
// 'int __attribute__((vector_size(16))) __device__ *'  (AS1), param 2 AS3.
typedef int v4i_ __attribute__((vector_size(16)));
typedef __attribute__((address_space(1))) v4i_ as1_v4i;
typedef __attribute__((address_space(3))) v4i_ as3_v4i;
__device__ __forceinline__ void gl_async_b128(const _Float16* g, _Float16* l) {
    // low 32 bits of a generic LDS pointer are the LDS byte address
    __builtin_amdgcn_global_load_async_to_lds_b128(
        (as1_v4i*)(uintptr_t)g, (as3_v4i*)(unsigned)(uintptr_t)l, 0, 0);
}
#else
#define HAVE_ASYNC_COPY 0
#endif

template <int N>
__device__ __forceinline__ void wait_async() {
#if HAVE_ASYNC_COPY
#if __has_builtin(__builtin_amdgcn_s_wait_asynccnt)
    __builtin_amdgcn_s_wait_asynccnt(N);
#else
    asm volatile("s_wait_asynccnt %0" ::"n"(N) : "memory");
#endif
#endif
}

// 32B (16 halfs) per thread: async issue OR register staging.
struct AStage { uint4 x, y; };
__device__ __forceinline__ void issueA(const _Float16* g, _Float16* l, AStage& st) {
#if HAVE_ASYNC_COPY
    gl_async_b128(g, l);
    gl_async_b128(g + 8, l + 8);
    (void)st;
#else
    st.x = *(const uint4*)g;
    st.y = *(const uint4*)(g + 8);
    (void)l;
#endif
}
__device__ __forceinline__ void commitA(_Float16* l, const AStage& st) {
#if !HAVE_ASYNC_COPY
    *(uint4*)l       = st.x;
    *(uint4*)(l + 8) = st.y;
#else
    (void)l; (void)st;
#endif
}

__device__ __forceinline__ v8f zero8() {
    v8f z = {0.f, 0.f, 0.f, 0.f, 0.f, 0.f, 0.f, 0.f};
    return z;
}

// A fragment: 16x32 f16, row-major LDS tile (row stride lda halfs).
__device__ __forceinline__ v16h frag_a(const _Float16* s, int lda, int lane) {
    const int m  = lane & 15;
    const int kb = (lane & 16) ? 8 : 0;
    const _Float16* p = s + m * lda + kb;
    v16h r;
#pragma unroll
    for (int i = 0; i < 8; ++i) { r[i] = p[i]; r[i + 8] = p[i + 16]; }
    return r;
}

// B fragment: 32x16 f16 from LDS stored TRANSPOSED as BsT[n][k].
__device__ __forceinline__ v16h frag_b(const _Float16* s, int ldb, int lane) {
    const int n  = lane & 15;
    const int kb = (lane & 16) ? 8 : 0;
    const _Float16* p = s + n * ldb + kb;
    v16h r;
#pragma unroll
    for (int i = 0; i < 8; ++i) { r[i] = p[i]; r[i + 8] = p[i + 16]; }
    return r;
}

__device__ __forceinline__ v8f wmma_f16(v16h a, v16h b, v8f c) {
    return __builtin_amdgcn_wmma_f32_16x16x32_f16(false, a, false, b, (short)0, c,
                                                  false, false);
}

// ---------------------------------------------------------------------------
__global__ void cvt_kernel(const float* __restrict__ src, _Float16* __restrict__ dst,
                           int n) {
    for (int i = blockIdx.x * blockDim.x + threadIdx.x; i < n;
         i += gridDim.x * blockDim.x)
        dst[i] = (_Float16)src[i];
}

// ---------------------------------------------------------------------------
// Tiled WMMA GEMM: C[M,N] = A[M,K](f16) @ B[K,N](f16), fp32 accum.
// 256 thr = 8 waves; tile 128(M) x 64(N); K-step 32; double-buffered LDS with
// async A-tile streaming one K-step ahead of the WMMA pipeline.
// ---------------------------------------------------------------------------
enum { EPI_F32 = 0, EPI_ELU1 = 1, EPI_SCALE = 2, EPI_RELU = 3 };

template <int EPI>
__global__ __launch_bounds__(256) void gemm_kernel(
    const _Float16* __restrict__ A, const _Float16* __restrict__ B,
    void* __restrict__ Cout, int M, int N, int K, float scale) {
    __shared__ __align__(16) _Float16 As[2][128 * 40];
    __shared__ __align__(16) _Float16 Bs[2][64 * 40];
    const int tid = threadIdx.x, lane = tid & 31, wid = tid >> 5;
    const int bm = blockIdx.y * 128, bn = blockIdx.x * 64;
    const int wm = (wid & 3) * 32, wn = (wid >> 2) * 32;
    const int ra = tid >> 1, ca = (tid & 1) * 16;  // A: 2 thr/row, 16 halfs each
    const int rb = tid >> 3, cb = (tid & 7) * 8;   // B: 8 thr/row, 8 halfs each
    const _Float16* Ag = A + (size_t)(bm + ra) * K + ca;
    const _Float16* Bg = B + (size_t)rb * N + bn + cb;
    const int aoff = ra * 40 + ca;
    const int nT = K >> 5;

    v8f acc[2][2];
#pragma unroll
    for (int i = 0; i < 2; ++i)
#pragma unroll
        for (int j = 0; j < 2; ++j) acc[i][j] = zero8();

    union Bu { uint4 u; _Float16 h[8]; };
    AStage sa, sn;
    Bu bv, nbv = {};
    issueA(Ag, &As[0][aoff], sa);                 // tile 0 in flight
    bv.u = *(const uint4*)Bg;

    for (int t = 0; t < nT; ++t) {
        const int cur = t & 1;
        __syncthreads();  // all compute(t-1) retired -> safe to fill buffers
        const bool more = (t + 1 < nT);
        if (more) {
            issueA(Ag + (t + 1) * 32, &As[cur ^ 1][aoff], sn);
            nbv.u = *(const uint4*)(Bg + (size_t)(t + 1) * 32 * N);
            if (t + 2 < nT)  // gfx1250 global_prefetch_b8, two tiles ahead
                __builtin_prefetch(Bg + (size_t)(t + 2) * 32 * N, 0, 1);
        }
        if (more) wait_async<2>(); else wait_async<0>();  // tile t landed in LDS
        commitA(&As[cur][aoff], sa);
#pragma unroll
        for (int i = 0; i < 8; ++i) Bs[cur][(cb + i) * 40 + rb] = bv.h[i];
        __syncthreads();  // tile t visible to all waves

        const v16h a0 = frag_a(&As[cur][(wm)      * 40], 40, lane);
        const v16h a1 = frag_a(&As[cur][(wm + 16) * 40], 40, lane);
        const v16h b0 = frag_b(&Bs[cur][(wn)      * 40], 40, lane);
        const v16h b1 = frag_b(&Bs[cur][(wn + 16) * 40], 40, lane);
        acc[0][0] = wmma_f16(a0, b0, acc[0][0]);
        acc[0][1] = wmma_f16(a0, b1, acc[0][1]);
        acc[1][0] = wmma_f16(a1, b0, acc[1][0]);
        acc[1][1] = wmma_f16(a1, b1, acc[1][1]);
        sa = sn;
        bv = nbv;
    }

    float*    C32 = (float*)Cout;
    _Float16* C16 = (_Float16*)Cout;
#pragma unroll
    for (int i = 0; i < 2; ++i)
#pragma unroll
        for (int j = 0; j < 2; ++j) {
            const int col = bn + wn + j * 16 + (lane & 15);
#pragma unroll
            for (int r = 0; r < 8; ++r) {
                const int row = bm + wm + i * 16 + r + ((lane & 16) >> 1);
                float v = acc[i][j][r];
                if (EPI == EPI_ELU1)  v = (v > 0.f) ? v + 1.f : __expf(v);
                if (EPI == EPI_SCALE) v = v * scale;
                if (EPI == EPI_RELU)  v = (v > 0.f) ? v : 0.f;
                if (EPI == EPI_F32) C32[(size_t)row * N + col] = v;
                else                C16[(size_t)row * N + col] = (_Float16)v;
            }
        }
}

// ---------------------------------------------------------------------------
// KV[b,h] = Kf[b,:,h,:]^T @ Vn[b,:,h,:]  (64x64, reduce over 4096 via WMMA)
// Chunks need a transpose into LDS, so this stays on the register path.
// ---------------------------------------------------------------------------
__global__ __launch_bounds__(256) void kv_kernel(
    const _Float16* __restrict__ Kf, const _Float16* __restrict__ Vn,
    _Float16* __restrict__ KV16) {
    __shared__ __align__(16) _Float16 KsT[64 * 40];  // A^T: [d][s_local]
    __shared__ __align__(16) _Float16 VsT[64 * 40];  // B^T: [v][s_local]
    const int tid = threadIdx.x, lane = tid & 31, wid = tid >> 5;
    const int bh = blockIdx.x;
    const size_t base = (size_t)(bh >> 3) * SEQ * D_MODEL + (size_t)(bh & 7) * HDIM;
    const int tm = (wid & 3) * 16, tn = (wid >> 2) * 32;
    const int si = tid >> 3, d0 = (tid & 7) * 8;
    v8f acc[2];
    acc[0] = zero8(); acc[1] = zero8();

    for (int s = 0; s < SEQ; s += 32) {
        union { uint4 u; _Float16 h[8]; } tk, tv;
        tk.u = *(const uint4*)(Kf + base + (size_t)(s + si) * D_MODEL + d0);
        tv.u = *(const uint4*)(Vn + base + (size_t)(s + si) * D_MODEL + d0);
        if (s + 32 < SEQ)
            __builtin_prefetch(Kf + base + (size_t)(s + 32 + si) * D_MODEL + d0, 0, 1);
        __syncthreads();
#pragma unroll
        for (int i = 0; i < 8; ++i) {
            KsT[(d0 + i) * 40 + si] = tk.h[i];
            VsT[(d0 + i) * 40 + si] = tv.h[i];
        }
        __syncthreads();
        const v16h a  = frag_a(&KsT[tm * 40], 40, lane);
        const v16h b0 = frag_b(&VsT[tn * 40], 40, lane);
        const v16h b1 = frag_b(&VsT[(tn + 16) * 40], 40, lane);
        acc[0] = wmma_f16(a, b0, acc[0]);
        acc[1] = wmma_f16(a, b1, acc[1]);
    }
#pragma unroll
    for (int j = 0; j < 2; ++j) {
        const int col = tn + j * 16 + (lane & 15);
#pragma unroll
        for (int r = 0; r < 8; ++r) {
            const int row = tm + r + ((lane & 16) >> 1);
            KV16[(size_t)bh * 4096 + row * 64 + col] = (_Float16)acc[j][r];
        }
    }
}

// Ksum[b,h,d] = sum_s Kf[b,s,h,d]
__global__ __launch_bounds__(256) void ksum_kernel(const _Float16* __restrict__ Kf,
                                                   float* __restrict__ Ksum) {
    __shared__ float part[256];
    const int tid = threadIdx.x, bh = blockIdx.x;
    const size_t base = (size_t)(bh >> 3) * SEQ * D_MODEL + (size_t)(bh & 7) * HDIM;
    const int d = tid & 63, g = tid >> 6;
    float s = 0.f;
    for (int i = g; i < SEQ; i += 4) s += (float)Kf[base + (size_t)i * D_MODEL + d];
    part[tid] = s;
    __syncthreads();
    if (tid < 64)
        Ksum[bh * 64 + tid] = part[tid] + part[tid + 64] + part[tid + 128] + part[tid + 192];
}

// Z[b,l,h] = 1 / (Qf[b,l,h,:] . Ksum[b,h,:] + eps)
__global__ __launch_bounds__(256) void z_kernel(const _Float16* __restrict__ Qf,
                                                const float* __restrict__ Ksum,
                                                float* __restrict__ Z) {
    const int idx = blockIdx.x * 256 + threadIdx.x;  // B*L*H = 131072
    const int h = idx & 7, bl = idx >> 3, b = bl >> 12;
    const _Float16* qp = Qf + (size_t)bl * D_MODEL + h * HDIM;
    const float* ks = Ksum + ((b << 3) + h) * HDIM;
    float dot = 0.f;
#pragma unroll 8
    for (int d = 0; d < HDIM; ++d) dot += (float)qp[d] * ks[d];
    Z[idx] = 1.f / (dot + 1e-6f);
}

// msg[b,l,h,:] = (Qf @ KV) * Z * S   -> f16, layout [16384, 512]
__global__ __launch_bounds__(256) void attn_kernel(
    const _Float16* __restrict__ Qf, const _Float16* __restrict__ KV16,
    const float* __restrict__ Z, _Float16* __restrict__ msg) {
    __shared__ __align__(16) _Float16 Qs[128 * 72];
    __shared__ __align__(16) _Float16 KVT[64 * 72];  // B^T: [v][d]
    __shared__ float zs[128];
    const int tid = threadIdx.x, lane = tid & 31, wid = tid >> 5;
    const int bh = blockIdx.x, b = bh >> 3, h = bh & 7;
    const int bm = blockIdx.y * 128;
    const int wm = (wid & 3) * 32, wn = (wid >> 2) * 32;
    {   // Q tile 128x64: transpose-free -> async straight into LDS
        const int row = tid >> 1, c0 = (tid & 1) * 32;
        const _Float16* gp =
            Qf + (size_t)(b * SEQ + bm + row) * D_MODEL + h * HDIM + c0;
        _Float16* lp = &Qs[row * 72 + c0];
#if HAVE_ASYNC_COPY
#pragma unroll
        for (int i = 0; i < 4; ++i) gl_async_b128(gp + i * 8, lp + i * 8);
#else
#pragma unroll
        for (int i = 0; i < 4; ++i)
            *(uint4*)(lp + i * 8) = *(const uint4*)(gp + i * 8);
#endif
    }
    {   // KV 64x64 -> transposed LDS (register path)
        const int d = tid >> 2, v0 = (tid & 3) * 16;
        union { uint4 u; _Float16 h8[8]; } t0, t1;
        t0.u = *(const uint4*)(KV16 + (size_t)bh * 4096 + d * 64 + v0);
        t1.u = *(const uint4*)(KV16 + (size_t)bh * 4096 + d * 64 + v0 + 8);
#pragma unroll
        for (int i = 0; i < 8; ++i) {
            KVT[(v0 + i) * 72 + d]     = t0.h8[i];
            KVT[(v0 + 8 + i) * 72 + d] = t1.h8[i];
        }
    }
    if (tid < 128)
        zs[tid] = Z[(size_t)(b * SEQ + bm + tid) * NHEAD + h] * (float)SEQ;
    wait_async<0>();
    __syncthreads();

    v8f acc[2][2];
#pragma unroll
    for (int i = 0; i < 2; ++i)
#pragma unroll
        for (int j = 0; j < 2; ++j) acc[i][j] = zero8();
#pragma unroll
    for (int kt = 0; kt < 64; kt += 32) {
        const v16h a0 = frag_a(&Qs[wm * 72 + kt], 72, lane);
        const v16h a1 = frag_a(&Qs[(wm + 16) * 72 + kt], 72, lane);
        const v16h b0 = frag_b(&KVT[wn * 72 + kt], 72, lane);
        const v16h b1 = frag_b(&KVT[(wn + 16) * 72 + kt], 72, lane);
        acc[0][0] = wmma_f16(a0, b0, acc[0][0]);
        acc[0][1] = wmma_f16(a0, b1, acc[0][1]);
        acc[1][0] = wmma_f16(a1, b0, acc[1][0]);
        acc[1][1] = wmma_f16(a1, b1, acc[1][1]);
    }
#pragma unroll
    for (int i = 0; i < 2; ++i)
#pragma unroll
        for (int j = 0; j < 2; ++j) {
            const int col = wn + j * 16 + (lane & 15);
#pragma unroll
            for (int r = 0; r < 8; ++r) {
                const int lrow = wm + i * 16 + r + ((lane & 16) >> 1);
                const float v = acc[i][j][r] * zs[lrow];
                msg[(size_t)(b * SEQ + bm + lrow) * D_MODEL + h * HDIM + col] =
                    (_Float16)v;
            }
        }
}

// LayerNorm over 512 features: out = LN(X + R) * gamma + beta
template <bool HALFOUT>
__global__ __launch_bounds__(128) void ln_kernel(
    const float* __restrict__ X, const float* __restrict__ R,
    const float* __restrict__ gamma, const float* __restrict__ beta,
    void* __restrict__ out) {
    __shared__ float red[128];
    __shared__ float s_mu, s_rs;
    const int t = threadIdx.x;
    const size_t base = (size_t)blockIdx.x * D_MODEL;
    const float4 x4 = ((const float4*)(X + base))[t];
    const float4 r4 = ((const float4*)(R + base))[t];
    const float v0 = x4.x + r4.x, v1 = x4.y + r4.y, v2 = x4.z + r4.z, v3 = x4.w + r4.w;
    red[t] = v0 + v1 + v2 + v3;
    __syncthreads();
    for (int off = 64; off > 0; off >>= 1) {
        if (t < off) red[t] += red[t + off];
        __syncthreads();
    }
    if (t == 0) s_mu = red[0] * (1.f / D_MODEL);
    __syncthreads();
    const float mu = s_mu;
    const float d0 = v0 - mu, d1 = v1 - mu, d2 = v2 - mu, d3 = v3 - mu;
    red[t] = d0 * d0 + d1 * d1 + d2 * d2 + d3 * d3;
    __syncthreads();
    for (int off = 64; off > 0; off >>= 1) {
        if (t < off) red[t] += red[t + off];
        __syncthreads();
    }
    if (t == 0) s_rs = rsqrtf(red[0] * (1.f / D_MODEL) + 1e-5f);
    __syncthreads();
    const float rs = s_rs;
    const float4 g4 = ((const float4*)gamma)[t];
    const float4 b4 = ((const float4*)beta)[t];
    const float y0 = d0 * rs * g4.x + b4.x;
    const float y1 = d1 * rs * g4.y + b4.y;
    const float y2 = d2 * rs * g4.z + b4.z;
    const float y3 = d3 * rs * g4.w + b4.w;
    if (HALFOUT) {
        _Float16* o = (_Float16*)out;
        o[base + t * 4 + 0] = (_Float16)y0;
        o[base + t * 4 + 1] = (_Float16)y1;
        o[base + t * 4 + 2] = (_Float16)y2;
        o[base + t * 4 + 3] = (_Float16)y3;
    } else {
        ((float4*)out)[blockIdx.x * 128 + t] = make_float4(y0, y1, y2, y3);
    }
}

// ---------------------------------------------------------------------------
extern "C" void kernel_launch(void* const* d_in, const int* in_sizes, int n_in,
                              void* d_out, int out_size, void* d_ws, size_t ws_size,
                              hipStream_t stream) {
    (void)in_sizes; (void)n_in; (void)out_size; (void)ws_size;
    const float* q  = (const float*)d_in[0];
    const float* k  = (const float*)d_in[1];
    const float* v  = (const float*)d_in[2];
    const float* Wq = (const float*)d_in[3];
    const float* Wk = (const float*)d_in[4];
    const float* Wv = (const float*)d_in[5];
    const float* Wm = (const float*)d_in[6];
    const float* W1 = (const float*)d_in[7];
    const float* W2 = (const float*)d_in[8];
    const float* g1 = (const float*)d_in[9];
    const float* b1 = (const float*)d_in[10];
    const float* g2 = (const float*)d_in[11];
    const float* b2 = (const float*)d_in[12];
    float* out = (float*)d_out;

    char* ws = (char*)d_ws;
    size_t o = 0;
    auto take = [&](size_t bytes) -> char* {
        char* p = ws + o;
        o += (bytes + 255) & ~(size_t)255;
        return p;
    };
    const size_t HB = (size_t)MROWS * D_MODEL * 2;  // 16.78 MB f16 buffer
    _Float16* hq  = (_Float16*)take(HB);
    _Float16* hk  = (_Float16*)take(HB);
    _Float16* hv  = (_Float16*)take(HB);  // start of the 4*HB "mid" alias block
    _Float16* Qf  = (_Float16*)take(HB);
    _Float16* Kf  = (_Float16*)take(HB);
    _Float16* Vn  = (_Float16*)take(HB);
    _Float16* hWq = (_Float16*)take((size_t)512 * 512 * 2);
    _Float16* hWk = (_Float16*)take((size_t)512 * 512 * 2);
    _Float16* hWv = (_Float16*)take((size_t)512 * 512 * 2);
    _Float16* hWm = (_Float16*)take((size_t)512 * 512 * 2);
    _Float16* hW1 = (_Float16*)take((size_t)512 * 2048 * 2);
    _Float16* hW2 = (_Float16*)take((size_t)2048 * 512 * 2);
    float*    msgWm = (float*)take((size_t)MROWS * D_MODEL * 4);
    float*    Ksum  = (float*)take((size_t)BATCH * NHEAD * HDIM * 4);
    float*    Zb    = (float*)take((size_t)BATCH * SEQ * NHEAD * 4);
    _Float16* KV16  = (_Float16*)take((size_t)BATCH * NHEAD * HDIM * HDIM * 2);
    // lifetime-based aliases
    _Float16* msg16 = hq;     // hq consumed by Q-projection before attn writes
    _Float16* x1h   = hk;     // hk consumed by K-projection before LN1 writes
    _Float16* mid   = hv;     // [hv,Qf,Kf,Vn] = 4*HB = exactly 16384*2048*2 B
    float*    ffn   = msgWm;  // msgWm consumed by LN1 before W2 writes

    auto cvt = [&](const float* s, _Float16* d, int n) {
        int blocks = (n + 1023) / 1024;
        if (blocks > 4096) blocks = 4096;
        cvt_kernel<<<blocks, 256, 0, stream>>>(s, d, n);
    };
    cvt(q, hq, MROWS * D_MODEL);
    cvt(k, hk, MROWS * D_MODEL);
    cvt(v, hv, MROWS * D_MODEL);
    cvt(Wq, hWq, 512 * 512);
    cvt(Wk, hWk, 512 * 512);
    cvt(Wv, hWv, 512 * 512);
    cvt(Wm, hWm, 512 * 512);
    cvt(W1, hW1, 512 * 2048);
    cvt(W2, hW2, 2048 * 512);

    const dim3 blk(256);
    // Q/K/V projections with fused feature-map epilogues
    gemm_kernel<EPI_ELU1><<<dim3(8, 128), blk, 0, stream>>>(hq, hWq, Qf, MROWS, 512, 512, 0.f);
    gemm_kernel<EPI_ELU1><<<dim3(8, 128), blk, 0, stream>>>(hk, hWk, Kf, MROWS, 512, 512, 0.f);
    gemm_kernel<EPI_SCALE><<<dim3(8, 128), blk, 0, stream>>>(hv, hWv, Vn, MROWS, 512, 512, 1.f / SEQ);
    // linear attention core
    ksum_kernel<<<BATCH * NHEAD, blk, 0, stream>>>(Kf, Ksum);
    kv_kernel<<<BATCH * NHEAD, blk, 0, stream>>>(Kf, Vn, KV16);
    z_kernel<<<(BATCH * SEQ * NHEAD) / 256, blk, 0, stream>>>(Qf, Ksum, Zb);
    attn_kernel<<<dim3(BATCH * NHEAD, SEQ / 128), blk, 0, stream>>>(Qf, KV16, Zb, msg16);
    // merge projection + LN1
    gemm_kernel<EPI_F32><<<dim3(8, 128), blk, 0, stream>>>(msg16, hWm, msgWm, MROWS, 512, 512, 0.f);
    ln_kernel<true><<<MROWS, 128, 0, stream>>>(msgWm, q, g1, b1, x1h);
    // FFN + LN2 (residual from original q)
    gemm_kernel<EPI_RELU><<<dim3(32, 128), blk, 0, stream>>>(x1h, hW1, mid, MROWS, 2048, 512, 0.f);
    gemm_kernel<EPI_F32><<<dim3(8, 128), blk, 0, stream>>>(mid, hW2, ffn, MROWS, 512, 2048, 0.f);
    ln_kernel<false><<<MROWS, 128, 0, stream>>>(ffn, q, g2, b2, out);
}
